// CantorAttentionPlus_58007828300068
// MI455X (gfx1250) — compile-verified, matrix operand order checked by reading, not run on hardware
//
#include <hip/hip_runtime.h>

// ---------------------------------------------------------------------------
// CantorAttentionPlus for MI455X (gfx1250, wave32, WMMA)
// Pipeline: f32->bf16 cvt | QKV GEMM (bf16 WMMA, f32 acc) | gather-attention |
//           out GEMM (bf16 WMMA, f32 acc) -> f32
// GEMM staging uses GLOBAL_LOAD_ASYNC_TO_LDS_B128 (ASYNCcnt) when the
// toolchain exposes the builtin; falls back to load+ds_store otherwise.
// ---------------------------------------------------------------------------

#define BDIM   1024   // model dim
#define NTOK   4096   // B * N
#define NHEADS 16
#define HD     64     // head dim
#define SEQ    2048   // N
#define KR     64     // routes per query

// Probe for the gfx1250 async-copy builtins (compile-safe: falls back).
#if defined(__has_builtin)
#if __has_builtin(__builtin_amdgcn_global_load_async_to_lds_b128)
#define HAVE_ASYNC_LDS 1
#endif
#endif
#ifndef HAVE_ASYNC_LDS
#define HAVE_ASYNC_LDS 0
#endif

typedef __attribute__((ext_vector_type(16))) __bf16 v16bf;
typedef __attribute__((ext_vector_type(8)))  float  v8f;
typedef __attribute__((ext_vector_type(4)))  int    v4i;

union FragU { uint4 u[2]; v16bf v; };

__device__ __forceinline__ unsigned short f2bf(float f) {
  unsigned u = __float_as_uint(f);
  u += 0x7fffu + ((u >> 16) & 1u);          // round-to-nearest-even
  return (unsigned short)(u >> 16);
}
__device__ __forceinline__ float bf2f(unsigned short s) {
  return __uint_as_float(((unsigned)s) << 16);
}
__device__ __forceinline__ float dp2(unsigned a, unsigned b) {
  return bf2f((unsigned short)(a & 0xffffu)) * bf2f((unsigned short)(b & 0xffffu))
       + bf2f((unsigned short)(a >> 16))     * bf2f((unsigned short)(b >> 16));
}

#if HAVE_ASYNC_LDS
// Per-lane 16B async copy global -> LDS, tracked by ASYNCcnt.
__device__ __forceinline__ void async_ld_b128(const unsigned short* g,
                                              unsigned short* l) {
  __builtin_amdgcn_global_load_async_to_lds_b128(
      (__attribute__((address_space(1))) v4i*)g,
      (__attribute__((address_space(3))) v4i*)l, 0, 0);
}
#endif

// ---------------------------------------------------------------------------
// Elementwise f32 -> bf16 (vectorized x4)
// ---------------------------------------------------------------------------
__global__ void cvt_f32_bf16(const float* __restrict__ src,
                             unsigned short* __restrict__ dst, int n4) {
  int i = blockIdx.x * blockDim.x + threadIdx.x;
  if (i >= n4) return;
  float4 v = ((const float4*)src)[i];
  uint2 o;
  o.x = ((unsigned)f2bf(v.y) << 16) | f2bf(v.x);
  o.y = ((unsigned)f2bf(v.w) << 16) | f2bf(v.z);
  ((uint2*)dst)[i] = o;
}

// ---------------------------------------------------------------------------
// C(MxN) = A(MxK) * B(NxK)^T + bias  -- bf16 inputs, f32 accumulate.
// Block tile 128x128, 8 waves (wave32), each wave 32x64 = 2x4 WMMA 16x16x32.
// MODE 0: store f32 row-major to Cout.
// MODE 1: scatter bf16 into (B,H,N,D)-layout q/k/v buffers (QKV projection).
// ---------------------------------------------------------------------------
template <int MODE>
__global__ __launch_bounds__(256) void gemm_bf16_wmma(
    const unsigned short* __restrict__ A,
    const unsigned short* __restrict__ Bm,
    const float* __restrict__ bias,
    float* __restrict__ Cout,
    unsigned short* __restrict__ qb,
    unsigned short* __restrict__ kb,
    unsigned short* __restrict__ vb,
    int Kdim, int Ncols)
{
  // 32 bf16 per row + 8 bf16 pad -> 80-byte row stride (16B aligned)
  __shared__ unsigned short As[128 * 40];
  __shared__ unsigned short Bs[128 * 40];

  const int tid  = threadIdx.x;
  const int bm   = blockIdx.y * 128;
  const int bn   = blockIdx.x * 128;
  const int wid  = tid >> 5, lane = tid & 31;
  const int wm   = (wid & 3) * 32;
  const int wn   = (wid >> 2) * 64;
  const int lrow = lane & 15;
  const int lkb  = (lane >> 4) * 8;   // first K chunk per ISA 16-bit A layout
  const int mofs = (lane >> 4) * 8;   // C-fragment row offset

  const v8f vzero = {0.f, 0.f, 0.f, 0.f, 0.f, 0.f, 0.f, 0.f};
  v8f acc[2][4];
  #pragma unroll
  for (int mt = 0; mt < 2; ++mt)
    #pragma unroll
    for (int nt = 0; nt < 4; ++nt)
      acc[mt][nt] = vzero;

  const int r0 = tid >> 2;        // staging: 64 rows / pass, 4x16B chunks/row
  const int c0 = (tid & 3) * 8;   // bf16 column of this 16B chunk

  for (int kk = 0; kk < Kdim; kk += 32) {
    #pragma unroll
    for (int half = 0; half < 2; ++half) {
      const int row = r0 + half * 64;
      const unsigned short* gA = A  + (size_t)(bm + row) * Kdim + kk + c0;
      const unsigned short* gB = Bm + (size_t)(bn + row) * Kdim + kk + c0;
      unsigned short* lA = &As[row * 40 + c0];
      unsigned short* lB = &Bs[row * 40 + c0];
#if HAVE_ASYNC_LDS
      // Async copy straight into LDS: no VGPR staging, tracked by ASYNCcnt.
      async_ld_b128(gA, lA);
      async_ld_b128(gB, lB);
#else
      uint4 va = *(const uint4*)gA;
      *(uint4*)lA = va;
      uint4 vw = *(const uint4*)gB;
      *(uint4*)lB = vw;
#endif
    }
    if (kk + 32 < Kdim) {   // pull next K-slab toward L2 (global_prefetch)
      __builtin_prefetch(A  + (size_t)(bm + r0) * Kdim + kk + 32 + c0, 0, 0);
      __builtin_prefetch(Bm + (size_t)(bn + r0) * Kdim + kk + 32 + c0, 0, 0);
    }
#if HAVE_ASYNC_LDS
#if __has_builtin(__builtin_amdgcn_s_wait_asynccnt)
    __builtin_amdgcn_s_wait_asynccnt(0);
#else
    asm volatile("s_wait_asynccnt 0" ::: "memory");
#endif
#endif
    __syncthreads();

    FragU af[2], bfr[4];
    #pragma unroll
    for (int mt = 0; mt < 2; ++mt) {
      const unsigned short* p = &As[(wm + mt * 16 + lrow) * 40 + lkb];
      af[mt].u[0] = *(const uint4*)p;
      af[mt].u[1] = *(const uint4*)(p + 16);
    }
    #pragma unroll
    for (int nt = 0; nt < 4; ++nt) {
      const unsigned short* p = &Bs[(wn + nt * 16 + lrow) * 40 + lkb];
      bfr[nt].u[0] = *(const uint4*)p;
      bfr[nt].u[1] = *(const uint4*)(p + 16);
    }
    #pragma unroll
    for (int mt = 0; mt < 2; ++mt)
      #pragma unroll
      for (int nt = 0; nt < 4; ++nt)
        acc[mt][nt] = __builtin_amdgcn_wmma_f32_16x16x32_bf16(
            false, af[mt].v, false, bfr[nt].v, (short)0, acc[mt][nt],
            false, false);
    __syncthreads();
  }

  #pragma unroll
  for (int mt = 0; mt < 2; ++mt) {
    #pragma unroll
    for (int nt = 0; nt < 4; ++nt) {
      const int col = bn + wn + nt * 16 + lrow;
      const float bv = bias[col];
      #pragma unroll
      for (int j = 0; j < 8; ++j) {
        const int m = bm + wm + mt * 16 + mofs + j;
        const float val = acc[mt][nt][j] + bv;
        if (MODE == 0) {
          Cout[(size_t)m * Ncols + col] = val;
        } else {
          const int which = col >> 10;          // 0=q 1=k 2=v
          const int h     = (col >> 6) & 15;
          const int d     = col & 63;
          const int bidx  = m >> 11;            // batch
          const int tok   = m & 2047;           // token
          unsigned short* dstp = (which == 0) ? qb : (which == 1) ? kb : vb;
          dstp[(((size_t)(bidx * NHEADS + h)) * SEQ + tok) * HD + d] = f2bf(val);
        }
      }
    }
  }
}

// ---------------------------------------------------------------------------
// Gathered attention: one wave32 per (b,h,query).
// Phase 1: lane -> 2 keys (128B-aligned row gathers), f32 dots, shfl softmax.
// Phase 2: lane -> 2 dims, all lanes stream the same V row (coalesced 128B).
// ---------------------------------------------------------------------------
__global__ __launch_bounds__(256) void cantor_attn(
    const unsigned short* __restrict__ qb,
    const unsigned short* __restrict__ kb,
    const unsigned short* __restrict__ vb,
    const int* __restrict__ routes,
    unsigned short* __restrict__ yb)
{
  __shared__ float sp[8][KR];
  __shared__ int   sr[8][KR];
  const int tid = threadIdx.x, wid = tid >> 5, lane = tid & 31;
  const int gq = blockIdx.x * 8 + wid;   // (b*H + h)*SEQ + n
  const int n  = gq & (SEQ - 1);
  const int bh = gq >> 11;

  // Query row (64 bf16 = 128B), kept packed in 32 dwords.
  unsigned qp[32];
  {
    const uint4* q4 = (const uint4*)(qb + (size_t)gq * HD);
    #pragma unroll
    for (int i = 0; i < 8; ++i) {
      uint4 t = q4[i];
      qp[4*i] = t.x; qp[4*i+1] = t.y; qp[4*i+2] = t.z; qp[4*i+3] = t.w;
    }
  }

  const int rA = routes[n * KR + 2 * lane];
  const int rB = routes[n * KR + 2 * lane + 1];
  const unsigned short* kbase = kb + (size_t)bh * SEQ * HD;
  const uint4* k0 = (const uint4*)(kbase + (size_t)rA * HD);
  const uint4* k1 = (const uint4*)(kbase + (size_t)rB * HD);

  float s0 = 0.f, s1 = 0.f;
  #pragma unroll
  for (int c = 0; c < 8; ++c) {
    uint4 ka = k0[c], kc = k1[c];
    s0 += dp2(ka.x, qp[4*c]) + dp2(ka.y, qp[4*c+1]) +
          dp2(ka.z, qp[4*c+2]) + dp2(ka.w, qp[4*c+3]);
    s1 += dp2(kc.x, qp[4*c]) + dp2(kc.y, qp[4*c+1]) +
          dp2(kc.z, qp[4*c+2]) + dp2(kc.w, qp[4*c+3]);
  }
  s0 *= 0.125f;  // 1/sqrt(64)
  s1 *= 0.125f;

  float mx = fmaxf(s0, s1);
  #pragma unroll
  for (int off = 16; off >= 1; off >>= 1)
    mx = fmaxf(mx, __shfl_xor(mx, off, 32));
  float p0 = __expf(s0 - mx), p1 = __expf(s1 - mx);
  float sum = p0 + p1;
  #pragma unroll
  for (int off = 16; off >= 1; off >>= 1)
    sum += __shfl_xor(sum, off, 32);
  const float inv = 1.0f / sum;

  sp[wid][2 * lane]     = p0 * inv;
  sp[wid][2 * lane + 1] = p1 * inv;
  sr[wid][2 * lane]     = rA;
  sr[wid][2 * lane + 1] = rB;
  __syncthreads();

  const unsigned short* vbase = vb + (size_t)bh * SEQ * HD;
  float a0 = 0.f, a1 = 0.f;
  #pragma unroll 8
  for (int k = 0; k < KR; ++k) {
    const float p = sp[wid][k];
    const int   r = sr[wid][k];
    const unsigned vv = *(const unsigned*)(vbase + (size_t)r * HD + 2 * lane);
    a0 += p * bf2f((unsigned short)(vv & 0xffffu));
    a1 += p * bf2f((unsigned short)(vv >> 16));
  }

  const int b = bh >> 4, h = bh & 15;
  const size_t yoff = ((size_t)(b * SEQ + n)) * BDIM + h * HD + 2 * lane;
  const unsigned outv = ((unsigned)f2bf(a1) << 16) | f2bf(a0);
  *(unsigned*)(yb + yoff) = outv;
}

// ---------------------------------------------------------------------------
extern "C" void kernel_launch(void* const* d_in, const int* in_sizes, int n_in,
                              void* d_out, int out_size, void* d_ws, size_t ws_size,
                              hipStream_t stream) {
  const float* x      = (const float*)d_in[0];
  const int*   routes = (const int*)  d_in[1];
  const float* w_qkv  = (const float*)d_in[2];
  const float* b_qkv  = (const float*)d_in[3];
  const float* w_out  = (const float*)d_in[4];
  const float* b_out  = (const float*)d_in[5];
  float* out = (float*)d_out;

  char* ws = (char*)d_ws;
  const size_t MB = (size_t)1 << 20;
  unsigned short* xb    = (unsigned short*)(ws + 0  * MB);  // 8 MiB
  unsigned short* wqkvb = (unsigned short*)(ws + 8  * MB);  // 6 MiB
  unsigned short* woutb = (unsigned short*)(ws + 14 * MB);  // 2 MiB
  unsigned short* qbuf  = (unsigned short*)(ws + 16 * MB);  // 8 MiB (B,H,N,D)
  unsigned short* kbuf  = (unsigned short*)(ws + 24 * MB);  // 8 MiB
  unsigned short* vbuf  = (unsigned short*)(ws + 32 * MB);  // 8 MiB
  unsigned short* ybuf  = (unsigned short*)(ws + 40 * MB);  // 8 MiB (NTOK,DIM)

  // f32 -> bf16 staging
  cvt_f32_bf16<<<4096, 256, 0, stream>>>(x,     xb,    NTOK * BDIM / 4);
  cvt_f32_bf16<<<3072, 256, 0, stream>>>(w_qkv, wqkvb, 3 * BDIM * BDIM / 4);
  cvt_f32_bf16<<<1024, 256, 0, stream>>>(w_out, woutb, BDIM * BDIM / 4);

  // QKV projection: (4096x3072) = xb(4096x1024) * wqkvb^T, scatter to q/k/v
  gemm_bf16_wmma<1><<<dim3(24, 32), 256, 0, stream>>>(
      xb, wqkvb, b_qkv, nullptr, qbuf, kbuf, vbuf, BDIM, 3 * BDIM);

  // Gathered attention: 65536 queries, 8 per block (one wave each)
  cantor_attn<<<NTOK * NHEADS / 8, 256, 0, stream>>>(
      qbuf, kbuf, vbuf, routes, ybuf);

  // Output projection: (4096x1024) = ybuf * woutb^T + b_out -> f32
  gemm_bf16_wmma<0><<<dim3(8, 32), 256, 0, stream>>>(
      ybuf, woutb, b_out, out, nullptr, nullptr, nullptr, BDIM, BDIM);
}
